// GraphLayer_plus_3513283248940
// MI455X (gfx1250) — matrix operand order repaired and checked by reading, not compiled, not run on hardware
//
#include <hip/hip_runtime.h>
#include <stdint.h>

typedef __attribute__((ext_vector_type(16))) _Float16 v16h;
typedef __attribute__((ext_vector_type(8)))  float    v8f;

#define B_   4
#define C_   64
#define N_   4096
#define KNN  20
#define NEG_INF (-3.0e38f)

// ---------------- gfx1250 async global->LDS copy (ASYNCcnt path) ------------
#if __has_builtin(__builtin_amdgcn_global_load_async_to_lds_b128)
#define HAVE_ASYNC_LDS 1
#endif

typedef int v4i_ __attribute__((vector_size(16)));
typedef __attribute__((address_space(1))) v4i_* gv4p;
typedef __attribute__((address_space(3))) v4i_* lv4p;

__device__ __forceinline__ void async_ldcpy16(void* lds, const void* g) {
#ifdef HAVE_ASYNC_LDS
  __builtin_amdgcn_global_load_async_to_lds_b128((gv4p)g, (lv4p)lds, 0, 0);
#else
  *(uint4*)lds = *(const uint4*)g;     // fallback: VGPR round-trip
#endif
}
__device__ __forceinline__ void async_wait() {
#ifdef HAVE_ASYNC_LDS
#if __has_builtin(__builtin_amdgcn_s_wait_asynccnt)
  __builtin_amdgcn_s_wait_asynccnt(0);
#else
  asm volatile("s_wait_asynccnt 0x0" ::: "memory");
#endif
#endif
}

// sorted-descending register top-k insert; fully unrolled so arrays stay in VGPRs
__device__ __forceinline__ void topk_insert(float (&v)[KNN], int (&id)[KNN],
                                            float nv, int nid) {
  if (nv <= v[KNN - 1]) return;
  v[KNN - 1] = nv; id[KNN - 1] = nid;
#pragma unroll
  for (int s = KNN - 1; s > 0; --s) {
    if (v[s] > v[s - 1]) {
      float tv = v[s]; v[s] = v[s - 1]; v[s - 1] = tv;
      int   ti = id[s]; id[s] = id[s - 1]; id[s - 1] = ti;
    }
  }
}

// ---------------- K0: squared norms + f16 / f32 N-major transposes ----------
__global__ __launch_bounds__(256) void prep_kernel(const float* __restrict__ x,
                                                   _Float16* __restrict__ xh,
                                                   float* __restrict__ xtf,
                                                   float* __restrict__ sq) {
  int gid = blockIdx.x * 256 + threadIdx.x;          // = b*N + n
  int bb = gid / N_, n = gid - bb * N_;
  const float* xb = x + (size_t)bb * C_ * N_ + n;
  float s = 0.f;
#pragma unroll 8
  for (int c = 0; c < C_; ++c) {
    float v = xb[(size_t)c * N_];
    s += v * v;
    xh [(size_t)gid * C_ + c] = (_Float16)v;
    xtf[(size_t)gid * C_ + c] = v;
  }
  sq[gid] = s;
}

// ------------- K1: fused WMMA pairwise-distance + masked top-20 -------------
// one workgroup = 16 rows of one batch; 4 waves, each wave owns a 16-col
// WMMA subtile of the 64-column stripe being processed.
__global__ __launch_bounds__(128) void dist_topk_kernel(
    const _Float16* __restrict__ xh, const float* __restrict__ sq,
    const int* __restrict__ localIdx, int* __restrict__ knnIdx,
    float* __restrict__ idxOutF) {
  __shared__ __align__(16) _Float16 As[16][72];   // 144B rows: B128-aligned,
  __shared__ __align__(16) _Float16 Bs[64][72];   // conflict-free frag reads
  __shared__ float    S[16][64];
  __shared__ float    sqj[64];
  __shared__ int      mloc[64];
  __shared__ float    sqi[16];
  __shared__ float    candV[16][8][KNN];
  __shared__ int      candI[16][8][KNN];

  const int tid  = threadIdx.x;
  const int lane = tid & 31, wv = tid >> 5;
  const int bb = blockIdx.x / (N_ / 16);
  const int i0 = (blockIdx.x % (N_ / 16)) * 16;
  const _Float16* xb = xh + (size_t)bb * N_ * C_;

  // stage A tile (16x64 f16 = 128 x 16B chunks) via async-to-LDS
  for (int t = tid; t < 16 * 8; t += 128) {
    int r = t >> 3, c8 = t & 7;
    async_ldcpy16(&As[r][c8 * 8], xb + (size_t)(i0 + r) * C_ + c8 * 8);
  }
  if (tid < 16) sqi[tid] = sq[bb * N_ + i0 + tid];
  async_wait();
  __syncthreads();

  // A fragments (K = 0..31 and 32..63), constant across all column stripes
  const int m = lane & 15, hf = lane >> 4;
  v16h a0, a1;
#pragma unroll
  for (int q = 0; q < 8; ++q) {
    int kb = (q < 4) ? (q * 2 + hf * 8) : (16 + (q - 4) * 2 + hf * 8);
    a0[2 * q]     = As[m][kb];
    a0[2 * q + 1] = As[m][kb + 1];
    a1[2 * q]     = As[m][kb + 32];
    a1[2 * q + 1] = As[m][kb + 33];
  }

  float bv[KNN]; int bi[KNN];
#pragma unroll
  for (int q = 0; q < KNN; ++q) { bv[q] = NEG_INF; bi[q] = 0; }

  const int scanRow = tid >> 3, scanSub = tid & 7;

  for (int j0 = 0; j0 < N_; j0 += 64) {
    // prefetch the stripe after next (gfx1250 global_prefetch_b8)
    if (j0 + 128 < N_) {
      const char* nb = (const char*)(xb + (size_t)(j0 + 128) * C_);
      __builtin_prefetch(nb + tid * 64, 0, 1);
    }
    // stage B tile (64x64 f16 = 512 x 16B chunks) via async-to-LDS
    for (int t = tid; t < 64 * 8; t += 128) {
      int r = t >> 3, c8 = t & 7;
      async_ldcpy16(&Bs[r][c8 * 8], xb + (size_t)(j0 + r) * C_ + c8 * 8);
    }
    if (tid < 64) {
      sqj[tid]  = sq[bb * N_ + j0 + tid];
      mloc[tid] = localIdx[bb * N_ + j0 + tid];
    }
    async_wait();
    __syncthreads();

    // this wave's 16x16 inner-product tile: two chained 16x16x32 f16 WMMAs
    const int n  = lane & 15;
    const int jb = wv * 16;
    const int kh = hf * 8;
    v16h b0, b1;
#pragma unroll
    for (int q = 0; q < 8; ++q) {
      b0[2 * q]     = Bs[jb + n][kh + 2 * q + (q >= 4 ? 8 : 0)];
      b0[2 * q + 1] = Bs[jb + n][kh + 2 * q + (q >= 4 ? 8 : 0) + 1];
      b1[2 * q]     = Bs[jb + n][32 + kh + 2 * q + (q >= 4 ? 8 : 0)];
      b1[2 * q + 1] = Bs[jb + n][32 + kh + 2 * q + (q >= 4 ? 8 : 0) + 1];
    }
    v8f acc = {};
    acc = __builtin_amdgcn_wmma_f32_16x16x32_f16(false, a0, false, b0,
                                                 (short)0, acc, false, false);
    acc = __builtin_amdgcn_wmma_f32_16x16x32_f16(false, a1, false, b1,
                                                 (short)0, acc, false, false);

    // pd = 2*inner - |xi|^2 - |xj|^2 ; non-local -> -inf (rank-equivalent)
#pragma unroll
    for (int r = 0; r < 8; ++r) {
      int row = r + 8 * hf;
      int col = jb + n;
      float pd = 2.f * acc[r] - sqi[row] - sqj[col];
      if (!mloc[col]) pd = NEG_INF;
      S[row][col] = pd;
    }
    __syncthreads();

    // threshold-gated register top-20 over this stripe (8 threads/row)
#pragma unroll
    for (int u = 0; u < 8; ++u) {
      int col = scanSub * 8 + u;
      topk_insert(bv, bi, S[scanRow][col], j0 + col);
    }
    __syncthreads();
  }

  // merge the 8 per-thread candidate lists of each row
#pragma unroll
  for (int q = 0; q < KNN; ++q) {
    candV[scanRow][scanSub][q] = bv[q];
    candI[scanRow][scanSub][q] = bi[q];
  }
  __syncthreads();
  if (tid < 16) {
    float fv[KNN]; int fi[KNN];
#pragma unroll
    for (int q = 0; q < KNN; ++q) { fv[q] = NEG_INF; fi[q] = 0; }
    for (int t = 0; t < 8; ++t)
      for (int q = 0; q < KNN; ++q)
        topk_insert(fv, fi, candV[tid][t][q], candI[tid][t][q]);
    size_t base = ((size_t)bb * N_ + (i0 + tid)) * KNN;
#pragma unroll
    for (int q = 0; q < KNN; ++q) {
      knnIdx[base + q]  = fi[q];
      idxOutF[base + q] = (float)(fi[q] + bb * N_);   // idx_flat output
    }
  }
}

// ----- K2: per-point 20x20 self-attention; mean over rows collapsed to -----
// ----- a single weighted neighbor sum (one wave per point)              -----
__global__ __launch_bounds__(128) void attn_kernel(
    const float* __restrict__ xtf, const int* __restrict__ knnIdx,
    float* __restrict__ outp) {
  __shared__ __align__(16) float knn[4][KNN][C_];
  __shared__ float sc[4][KNN][KNN];
  __shared__ int   sidx[4][KNN];
  __shared__ float wj[4][KNN];

  const int tid = threadIdx.x, lane = tid & 31, wv = tid >> 5;
  const int pt = blockIdx.x * 4 + wv;                // b*N + n
  const int bb = pt / N_;
  if (lane < KNN) sidx[wv][lane] = knnIdx[(size_t)pt * KNN + lane];
  __syncthreads();

  // gather 20 neighbor rows (20 x 256B = 320 x 16B chunks) async into LDS
  const float* xb = xtf + (size_t)bb * N_ * C_;
  for (int t = lane; t < KNN * 16; t += 32) {
    int mm = t >> 4, c4 = t & 15;
    int j = sidx[wv][mm];
    async_ldcpy16(&knn[wv][mm][c4 * 4], xb + (size_t)j * C_ + c4 * 4);
  }
  async_wait();
  __syncthreads();

  const float scale = 0.125f;                        // 1/sqrt(64)
  for (int t = 0; t < 13; ++t) {
    int e = t * 32 + lane;
    if (e < KNN * KNN) {
      int i = e / KNN, jj = e - i * KNN;
      float d = 0.f;
#pragma unroll 8
      for (int c = 0; c < C_; ++c) d += knn[wv][i][c] * knn[wv][jj][c];
      sc[wv][i][jj] = d * scale;
    }
  }
  __syncthreads();

  if (lane < KNN) {                                  // softmax over own row
    float mx = NEG_INF;
    for (int jj = 0; jj < KNN; ++jj) mx = fmaxf(mx, sc[wv][lane][jj]);
    float sm = 0.f;
    for (int jj = 0; jj < KNN; ++jj) {
      float e = __expf(sc[wv][lane][jj] - mx);
      sc[wv][lane][jj] = e;
      sm += e;
    }
    float inv = 1.f / sm;
    for (int jj = 0; jj < KNN; ++jj) sc[wv][lane][jj] *= inv;
  }
  __syncthreads();

  if (lane < KNN) {                                  // w_j = mean_i attn[i][j]
    float s = 0.f;
    for (int i = 0; i < KNN; ++i) s += sc[wv][i][lane];
    wj[wv][lane] = s * (1.f / KNN);
  }
  __syncthreads();

  float o0 = 0.f, o1 = 0.f;
  for (int jj = 0; jj < KNN; ++jj) {
    float w = wj[wv][jj];
    o0 += w * knn[wv][jj][lane];
    o1 += w * knn[wv][jj][lane + 32];
  }
  outp[(size_t)pt * C_ + lane]      = o0;
  outp[(size_t)pt * C_ + lane + 32] = o1;
}

// ------ K3: edge-feature scatter (B,2C,N,k); contiguous store runs ----------
__global__ __launch_bounds__(320) void edge_kernel(
    const float* __restrict__ x, const float* __restrict__ outp,
    const int* __restrict__ knnIdx, float* __restrict__ feat) {
  const int tid = threadIdx.x;                       // 16 points * 20 nbrs
  const int bb = blockIdx.x / (N_ / 16);
  const int n0 = (blockIdx.x % (N_ / 16)) * 16;
  const int nl = tid / KNN, m = tid - nl * KNN;
  const int n = n0 + nl;
  const int j = knnIdx[((size_t)bb * N_ + n) * KNN + m];
  const float* ob = outp + ((size_t)bb * N_ + j) * C_;
  const size_t xrow  = (size_t)bb * C_ * N_;
  const size_t obase = (size_t)bb * 2 * C_ * N_ * KNN;
#pragma unroll 4
  for (int ch = 0; ch < C_; ++ch) {
    float xr = x[xrow + (size_t)ch * N_ + n];
    feat[obase + ((size_t)ch * N_ + n) * KNN + m]        = ob[ch] - xr;  // nb - x
    feat[obase + ((size_t)(ch + C_) * N_ + n) * KNN + m] = xr;           // x
  }
}

extern "C" void kernel_launch(void* const* d_in, const int* in_sizes, int n_in,
                              void* d_out, int out_size, void* d_ws, size_t ws_size,
                              hipStream_t stream) {
  const float* x        = (const float*)d_in[0];
  const int*   localIdx = (const int*)d_in[1];
  (void)in_sizes; (void)n_in; (void)out_size; (void)ws_size;   // k fixed = 20

  char* ws = (char*)d_ws;
  size_t off = 0;
  _Float16* xh = (_Float16*)(ws + off); off += (size_t)B_ * N_ * C_ * 2;  off = (off + 255) & ~(size_t)255;
  float* xtf   = (float*)(ws + off);    off += (size_t)B_ * N_ * C_ * 4;  off = (off + 255) & ~(size_t)255;
  float* sq    = (float*)(ws + off);    off += (size_t)B_ * N_ * 4;       off = (off + 255) & ~(size_t)255;
  int*   knnIdx= (int*)(ws + off);      off += (size_t)B_ * N_ * KNN * 4; off = (off + 255) & ~(size_t)255;
  float* outp  = (float*)(ws + off);

  float* feat    = (float*)d_out;
  float* idxOutF = feat + (size_t)B_ * 2 * C_ * N_ * KNN;  // tuple out #2

  prep_kernel     <<<B_ * N_ / 256, 256, 0, stream>>>(x, xh, xtf, sq);
  dist_topk_kernel<<<B_ * (N_ / 16), 128, 0, stream>>>(xh, sq, localIdx, knnIdx, idxOutF);
  attn_kernel     <<<B_ * N_ / 4,   128, 0, stream>>>(xtf, knnIdx, outp);
  edge_kernel     <<<B_ * (N_ / 16), 320, 0, stream>>>(x, outp, knnIdx, feat);
}
